// GCN_E_46969762349346
// MI455X (gfx1250) — matrix operand order repaired and checked by reading, not compiled, not run on hardware
//
#include <hip/hip_runtime.h>

// ---------------------------------------------------------------------------
// Types for CDNA5 WMMA (wave32): v_wmma_f32_16x16x32_bf16
// ---------------------------------------------------------------------------
typedef __attribute__((ext_vector_type(16))) __bf16          bf16x16;
typedef __attribute__((ext_vector_type(8)))  float           f32x8;
typedef __attribute__((ext_vector_type(16))) unsigned short  u16x16;
typedef __attribute__((ext_vector_type(8)))  unsigned short  u16x8;

__device__ __forceinline__ unsigned short f32_to_bf16(float f) {
    unsigned int u = __builtin_bit_cast(unsigned int, f);
    unsigned int r = u + 0x7fffu + ((u >> 16) & 1u);   // round-to-nearest-even
    return (unsigned short)(r >> 16);
}
__device__ __forceinline__ float bf16_to_f32(unsigned short h) {
    unsigned int u = ((unsigned int)h) << 16;
    return __builtin_bit_cast(float, u);
}

// Build one 16-half WMMA fragment from two contiguous 128-bit global loads.
// hiOff is in halfword units (A fragment: +8, B fragment: +8).
__device__ __forceinline__ bf16x16 load_frag(const unsigned short* __restrict__ p,
                                             int hiOff) {
    const u16x8 lo = __builtin_bit_cast(u16x8, *(const uint4*)(p));
    const u16x8 hi = __builtin_bit_cast(u16x8, *(const uint4*)(p + hiOff));
    const u16x16 f = __builtin_shufflevector(lo, hi,
        0,1,2,3,4,5,6,7,8,9,10,11,12,13,14,15);
    return __builtin_bit_cast(bf16x16, f);
}

// ---------------------------------------------------------------------------
// Elementwise f32 -> bf16 cast
// ---------------------------------------------------------------------------
__global__ void cast_f32_to_bf16(const float* __restrict__ in,
                                 unsigned short* __restrict__ out, int n) {
    int i = (blockIdx.x * blockDim.x + threadIdx.x) * 4;
    if (i + 4 <= n) {
        const float4 v = *(const float4*)(in + i);
        out[i + 0] = f32_to_bf16(v.x);
        out[i + 1] = f32_to_bf16(v.y);
        out[i + 2] = f32_to_bf16(v.z);
        out[i + 3] = f32_to_bf16(v.w);
    } else {
        for (int j = i; j < n; ++j) out[j] = f32_to_bf16(in[j]);
    }
}

// ---------------------------------------------------------------------------
// Batched cast + transpose: in f32 [Bc][R][Cc] -> out bf16 [Bc][Cc][R]
// (used once for x -> x^T and for the small weight matrices)
// ---------------------------------------------------------------------------
__global__ void cast_transpose_f32_bf16(const float* __restrict__ in,
                                        unsigned short* __restrict__ out,
                                        int Bc, int R, int Cc) {
    const long long total = (long long)Bc * R * Cc;
    const long long o = (long long)blockIdx.x * blockDim.x + threadIdx.x;
    if (o >= total) return;
    const long long rc  = (long long)R * Cc;
    const int b   = (int)(o / rc);
    const int rem = (int)(o - (long long)b * rc);
    const int c   = rem / R;            // output row   = input col
    const int r   = rem - c * R;        // output col   = input row
    out[o] = f32_to_bf16(in[(long long)b * rc + (long long)r * Cc + c]);
}

// ---------------------------------------------------------------------------
// LDS-free WMMA GEMM:  C = act( A0@B0 [+ A1@B1] + bias )
//   A  : bf16 row-major [M][K]        (fragment = 2x global_load_b128)
//   BT : bf16 row-major [N][K] (B^T)  (fragment = 2x global_load_b128)
//   C  : bf16 row-major [M][ldc]
//   CT : optional bf16 batch-blocked transpose [m/1024][ldc][1024]
//        (one packed b128 store per lane -- feeds next layer's bmm B^T)
//   Block = 256 threads = 8 waves (4M x 2N), wave tile 32 x 64
//   (2 A frags x 4 B frags -> 8 WMMAs per 12 b128 loads), block tile 128x128,
//   K-step 32, gridDim.z = batch (64-bit strides).
//   NSRC: 1|2 source pairs (dual-source = split of cat([h,agg]) @ W)
//   ACT : 0=none, 1=relu, 2=prelu(alpha per output channel)
//   B pointers are individual named registers (not an array) so they stay in
//   the global address space -> global_load_b128 (LOADcnt only), not flat.
// ---------------------------------------------------------------------------
template<int NSRC, int ACT, bool WRITE_T>
__global__ __launch_bounds__(256)
void wmma_gemm_kernel(const unsigned short* __restrict__ A0,
                      const unsigned short* __restrict__ BT0,
                      const unsigned short* __restrict__ A1,
                      const unsigned short* __restrict__ BT1,
                      unsigned short* __restrict__ C,
                      unsigned short* __restrict__ CT,
                      const float* __restrict__ bias,
                      const float* __restrict__ alpha,
                      int K, int lda, int ldbt, int ldc,
                      long long strideA, long long strideBT, long long strideC)
{
    const int lane = threadIdx.x & 31;
    const int wave = threadIdx.x >> 5;
    const int m    = lane & 15;      // row (A) / col (B,C) index within 16
    const int g    = lane >> 4;      // lane half-group

    const int m0 = blockIdx.y * 128 + (wave & 3) * 32;   // 4 M-waves
    const int n0 = blockIdx.x * 128 + (wave >> 2) * 64;  // 2 N-waves
    const long long zb = blockIdx.z;

    const f32x8 zero = {0.f,0.f,0.f,0.f,0.f,0.f,0.f,0.f};
    f32x8 acc[8] = {zero, zero, zero, zero, zero, zero, zero, zero}; // [mt*4+nt]

    for (int src = 0; src < NSRC; ++src) {
        const unsigned short* A  = ((src == 0) ? A0  : A1)  + zb * strideA;
        const unsigned short* BT = ((src == 0) ? BT0 : BT1) + zb * strideBT;

        // Per-lane fragment base pointers (contiguous k order, same for A and B):
        //  A 16x32 : lane holds row m, K = g*16 .. g*16+15 (contiguous in A)
        //  B 32x16 : lane holds col n, K = g*16 .. g*16+15 (contiguous in BT)
        const unsigned short* pA0 = A  + (long long)(m0 + m) * lda  + g * 16;
        const unsigned short* pA1 = pA0 + 16 * (long long)lda;
        const unsigned short* pB0 = BT + (long long)(n0 +  0 + m) * ldbt + g * 16;
        const unsigned short* pB1 = BT + (long long)(n0 + 16 + m) * ldbt + g * 16;
        const unsigned short* pB2 = BT + (long long)(n0 + 32 + m) * ldbt + g * 16;
        const unsigned short* pB3 = BT + (long long)(n0 + 48 + m) * ldbt + g * 16;

        for (int k = 0; k < K; k += 32) {
            const bf16x16 a0 = load_frag(pA0, 8);
            const bf16x16 a1 = load_frag(pA1, 8);
            const bf16x16 b0 = load_frag(pB0, 8);
            const bf16x16 b1 = load_frag(pB1, 8);
            const bf16x16 b2 = load_frag(pB2, 8);
            const bf16x16 b3 = load_frag(pB3, 8);

            acc[0] = __builtin_amdgcn_wmma_f32_16x16x32_bf16(
                false, a0, false, b0, (short)0, acc[0], false, false);
            acc[4] = __builtin_amdgcn_wmma_f32_16x16x32_bf16(
                false, a1, false, b0, (short)0, acc[4], false, false);
            acc[1] = __builtin_amdgcn_wmma_f32_16x16x32_bf16(
                false, a0, false, b1, (short)0, acc[1], false, false);
            acc[5] = __builtin_amdgcn_wmma_f32_16x16x32_bf16(
                false, a1, false, b1, (short)0, acc[5], false, false);
            acc[2] = __builtin_amdgcn_wmma_f32_16x16x32_bf16(
                false, a0, false, b2, (short)0, acc[2], false, false);
            acc[6] = __builtin_amdgcn_wmma_f32_16x16x32_bf16(
                false, a1, false, b2, (short)0, acc[6], false, false);
            acc[3] = __builtin_amdgcn_wmma_f32_16x16x32_bf16(
                false, a0, false, b3, (short)0, acc[3], false, false);
            acc[7] = __builtin_amdgcn_wmma_f32_16x16x32_bf16(
                false, a1, false, b3, (short)0, acc[7], false, false);

            pA0 += 32; pA1 += 32;
            pB0 += 32; pB1 += 32; pB2 += 32; pB3 += 32;
        }
    }

    // ---- epilogue: bias + activation; C/D layout: lane = col, vgpr r = row ----
    unsigned short* Cb = C + zb * strideC;
    #pragma unroll
    for (int mt = 0; mt < 2; ++mt) {
        #pragma unroll
        for (int nt = 0; nt < 4; ++nt) {
            const f32x8 a = acc[mt * 4 + nt];
            const int ncol = n0 + nt * 16 + m;
            const float bval = bias ? bias[ncol] : 0.f;
            const float aval = (ACT == 2) ? alpha[ncol] : 0.f;
            const int mbase = m0 + mt * 16 + g * 8;
            u16x8 pk;
            #pragma unroll
            for (int r = 0; r < 8; ++r) {
                float v = a[r] + bval;
                if (ACT == 1) v = (v > 0.f) ? v : 0.f;
                if (ACT == 2) v = (v > 0.f) ? v : aval * v;
                const unsigned short us = f32_to_bf16(v);
                Cb[(long long)(mbase + r) * ldc + ncol] = us;
                pk[r] = us;
            }
            if (WRITE_T) {
                // CT[b][ncol][node], node = mbase % 1024 (8 rows contiguous)
                const int bb   = mbase >> 10;
                const int node = mbase & 1023;
                unsigned short* p = CT + (long long)bb * ldc * 1024
                                       + (long long)ncol * 1024 + node;
                *(uint4*)p = __builtin_bit_cast(uint4, pk);
            }
        }
    }
}

// ---------------------------------------------------------------------------
// Tail: logits = hc @ Wc2 + bc2 (2 cols), then exact 2-class log_softmax (f32)
// ---------------------------------------------------------------------------
__global__ void classifier_tail(const unsigned short* __restrict__ hc,
                                const unsigned short* __restrict__ Wc2,
                                const float* __restrict__ bc2,
                                float* __restrict__ out, int rows)
{
    __shared__ float w0[256];
    __shared__ float w1[256];
    for (int d = threadIdx.x; d < 256; d += blockDim.x) {
        w0[d] = bf16_to_f32(Wc2[d * 2 + 0]);
        w1[d] = bf16_to_f32(Wc2[d * 2 + 1]);
    }
    __syncthreads();

    const int row = blockIdx.x * blockDim.x + threadIdx.x;
    if (row >= rows) return;
    const unsigned short* h = hc + (long long)row * 256;

    float l0 = 0.f, l1 = 0.f;
    for (int d = 0; d < 256; d += 8) {
        const uint4 p = *(const uint4*)(h + d);
        const unsigned int pk[4] = {p.x, p.y, p.z, p.w};
        #pragma unroll
        for (int j = 0; j < 4; ++j) {
            const float lo = bf16_to_f32((unsigned short)(pk[j] & 0xffffu));
            const float hi = bf16_to_f32((unsigned short)(pk[j] >> 16));
            l0 += lo * w0[d + j * 2] + hi * w0[d + j * 2 + 1];
            l1 += lo * w1[d + j * 2] + hi * w1[d + j * 2 + 1];
        }
    }
    l0 += bc2[0];
    l1 += bc2[1];
    const float mx  = fmaxf(l0, l1);
    const float lse = mx + logf(expf(l0 - mx) + expf(l1 - mx));
    out[(long long)row * 2 + 0] = l0 - lse;
    out[(long long)row * 2 + 1] = l1 - lse;
}

// ---------------------------------------------------------------------------
// Host orchestration
// ---------------------------------------------------------------------------
extern "C" void kernel_launch(void* const* d_in, const int* in_sizes, int n_in,
                              void* d_out, int out_size, void* d_ws, size_t ws_size,
                              hipStream_t stream) {
    (void)in_sizes; (void)n_in; (void)out_size; (void)ws_size;

    const float* x     = (const float*)d_in[0];
    const float* adj   = (const float*)d_in[1];
    const float* W1    = (const float*)d_in[2];
    const float* b1    = (const float*)d_in[3];
    const float* W2    = (const float*)d_in[4];
    const float* b2    = (const float*)d_in[5];
    const float* W3    = (const float*)d_in[6];
    const float* b3    = (const float*)d_in[7];
    const float* W4    = (const float*)d_in[8];
    const float* b4    = (const float*)d_in[9];
    const float* Wc1   = (const float*)d_in[10];
    const float* bc1   = (const float*)d_in[11];
    const float* alpha = (const float*)d_in[12];
    const float* Wc2   = (const float*)d_in[13];
    const float* bc2   = (const float*)d_in[14];
    float* out = (float*)d_out;

    const int B = 64, Nn = 1024;
    const int ROWS = B * Nn;                 // 65536 node rows

    // workspace carve-up (bf16 buffers), all offsets 256B aligned
    char* ws = (char*)d_ws;
    size_t off = 0;
    auto carve = [&](size_t elems) {
        unsigned short* p = (unsigned short*)(ws + off);
        off += elems * 2;
        return p;
    };
    unsigned short* adj_bf = carve((size_t)B * Nn * Nn);   // 128 MB, row-major
    unsigned short* h0     = carve((size_t)ROWS * 512);    // 64 MB ping (row-major)
    unsigned short* h1     = carve((size_t)ROWS * 512);    // 64 MB pong (row-major)
    unsigned short* aggb   = carve((size_t)ROWS * 512);    // 64 MB agg (row-major)
    unsigned short* hT     = carve((size_t)ROWS * 512);    // 64 MB h^T [B][D][N]
    unsigned short* w1T    = carve(512 * 512);             // [F][2Din] transposed
    unsigned short* w2T    = carve(512 * 1024);
    unsigned short* w3T    = carve(256 * 1024);
    unsigned short* w4T    = carve(256 * 512);
    unsigned short* wc1T   = carve(256 * 256);
    unsigned short* wc2bf  = carve(256 * 2);

    auto cast = [&](const float* src, unsigned short* dst, int n) {
        cast_f32_to_bf16<<<(n / 4 + 255) / 256, 256, 0, stream>>>(src, dst, n);
    };
    auto castT = [&](const float* src, unsigned short* dst, int Bc, int R, int Cc) {
        const long long total = (long long)Bc * R * Cc;
        cast_transpose_f32_bf16<<<(int)((total + 255) / 256), 256, 0, stream>>>(
            src, dst, Bc, R, Cc);
    };

    cast (adj, adj_bf, B * Nn * Nn);
    cast (x,   h0,     ROWS * 256);          // row-major x (layer-1 dense A)
    castT(x,   hT,     B, Nn, 256);          // x^T [B][256][N] (layer-1 bmm B^T)
    castT(W1,  w1T,    1, 512,  512);
    castT(W2,  w2T,    1, 1024, 512);
    castT(W3,  w3T,    1, 1024, 256);
    castT(W4,  w4T,    1, 512,  256);
    castT(Wc1, wc1T,   1, 256,  256);
    cast (Wc2, wc2bf,  256 * 2);

    // agg = adj @ h : per-batch, A=adj row-major, BT=h^T [D][N]
    auto bmm = [&](int D) {
        dim3 grid(D / 128, Nn / 128, B);
        wmma_gemm_kernel<1, 0, false><<<grid, 256, 0, stream>>>(
            adj_bf, hT, nullptr, nullptr, aggb, nullptr,
            nullptr, nullptr,
            /*K=*/Nn, /*lda=*/Nn, /*ldbt=*/Nn, /*ldc=*/D,
            (long long)Nn * Nn, (long long)D * Nn, (long long)Nn * D);
    };
    // h' = relu(h @ W_top + agg @ W_bot + b); writes h' and h'^T (next bmm B^T)
    auto gconv = [&](const unsigned short* hin, const unsigned short* WT,
                     const float* bias, unsigned short* hout, int Din, int F) {
        dim3 grid(F / 128, ROWS / 128, 1);
        wmma_gemm_kernel<2, 1, true><<<grid, 256, 0, stream>>>(
            hin, WT, aggb, WT + Din, hout, hT,
            bias, nullptr,
            /*K=*/Din, /*lda=*/Din, /*ldbt=*/2 * Din, /*ldc=*/F, 0, 0, 0);
    };

    bmm(256);  gconv(h0, w1T, b1, h1, 256, 512);   // layer 1: 256 -> 512
    bmm(512);  gconv(h1, w2T, b2, h0, 512, 512);   // layer 2: 512 -> 512
    bmm(512);  gconv(h0, w3T, b3, h1, 512, 256);   // layer 3: 512 -> 256
    bmm(256);  gconv(h1, w4T, b4, h0, 256, 256);   // layer 4: 256 -> 256

    // classifier: hc = prelu(h @ Wc1 + bc1), fused PReLU epilogue
    {
        dim3 grid(256 / 128, ROWS / 128, 1);
        wmma_gemm_kernel<1, 2, false><<<grid, 256, 0, stream>>>(
            h0, wc1T, nullptr, nullptr, h1, nullptr,
            bc1, alpha,
            256, 256, 256, 256, 0, 0, 0);
    }

    // logits + log_softmax
    classifier_tail<<<ROWS / 256, 256, 0, stream>>>(h1, wc2bf, bc2, out, ROWS);
}
